// ConvWindowAttention_45561013076014
// MI455X (gfx1250) — matrix operand order repaired and checked by reading, not compiled, not run on hardware
//
#include <hip/hip_runtime.h>

// ---------- types ----------
typedef __attribute__((ext_vector_type(16))) __bf16          v16bf;
typedef __attribute__((ext_vector_type(8)))  float           v8f;
typedef __attribute__((ext_vector_type(8)))  unsigned short  us8;
typedef __attribute__((ext_vector_type(4)))  unsigned short  us4;
typedef __attribute__((ext_vector_type(4)))  float           f4;

#define BK   64
#define LDB  72   // 64 + 8 pad bf16 elems -> 144B row stride: 16B aligned, conflict-free

__device__ __forceinline__ unsigned short f2bf(float f) {
  unsigned int u = __builtin_bit_cast(unsigned int, f);
  u += 0x7FFFu + ((u >> 16) & 1u);          // round-to-nearest-even
  return (unsigned short)(u >> 16);
}

union FragU { us8 h[2]; v16bf v; };

// ---------- one-pass f32 -> bf16 conversion (x, qkv_w, out_w) ----------
__global__ __launch_bounds__(256) void cvt_f32_bf16_kernel(
    const float* __restrict__ src, unsigned short* __restrict__ dst, int n4)
{
  int i = blockIdx.x * 256 + threadIdx.x;
  if (i < n4) {
    f4 v = reinterpret_cast<const f4*>(src)[i];
    us4 h = { f2bf(v.x), f2bf(v.y), f2bf(v.z), f2bf(v.w) };
    reinterpret_cast<us4*>(dst)[i] = h;
  }
}

// Issue async global->LDS copies for one 128x64 bf16 tile of A and of W.
// 256 threads x 4 iters x 16B = 16KB per matrix per stage.
__device__ __forceinline__ void stage_async(
    const unsigned short* __restrict__ gA, const unsigned short* __restrict__ gW,
    unsigned ldsA, unsigned ldsB, int tid, int K, int bm, int bn, int k0)
{
  #pragma unroll
  for (int it = 0; it < 4; ++it) {
    int idx = tid + it * 256;     // 0..1023
    int row = idx >> 3;           // 0..127
    int c8  = idx & 7;            // 16-byte chunk within 64-elem K slice
    unsigned long long ga =
        (unsigned long long)(gA + (size_t)(bm + row) * K + k0 + c8 * 8);
    unsigned long long gw =
        (unsigned long long)(gW + (size_t)(bn + row) * K + k0 + c8 * 8);
    unsigned la = ldsA + (unsigned)(row * (LDB * 2) + c8 * 16);
    unsigned lb = ldsB + (unsigned)(row * (LDB * 2) + c8 * 16);
    asm volatile("global_load_async_to_lds_b128 %0, %1, off"
                 :: "v"(la), "v"(ga) : "memory");
    asm volatile("global_load_async_to_lds_b128 %0, %1, off"
                 :: "v"(lb), "v"(gw) : "memory");
  }
}

// C[M,Nn] = A[M,K](bf16) * W[Nn,K](bf16)^T + bias(f32) ; f32 out.
// Block tile 128x128x64, 8 waves, wave tile 64x32 (4x2 WMMA tiles), 16 WMMA/stage,
// double-buffered LDS with ASYNCcnt-tracked global->LDS copies.
__global__ __launch_bounds__(256) void gemm_bf16_wmma_kernel(
    const unsigned short* __restrict__ A, const unsigned short* __restrict__ W,
    const float* __restrict__ bias, float* __restrict__ C,
    int M, int Nn, int K)
{
  __shared__ unsigned short As[2][128 * LDB];
  __shared__ unsigned short Bs[2][128 * LDB];

  const int tid  = threadIdx.x;
  const int lane = tid & 31;
  const int wave = tid >> 5;
  const int wm   = wave & 1;          // 64-row slab within block tile
  const int wn   = wave >> 1;         // 32-col slab within block tile
  const int bm   = blockIdx.y * 128;
  const int bn   = blockIdx.x * 128;
  const int half = lane >> 4;         // 0/1
  const int l16  = lane & 15;

  // LDS byte offsets (low 32 bits of generic LDS pointer == LDS offset)
  const unsigned asBase[2] = { (unsigned)(size_t)&As[0][0], (unsigned)(size_t)&As[1][0] };
  const unsigned bsBase[2] = { (unsigned)(size_t)&Bs[0][0], (unsigned)(size_t)&Bs[1][0] };

  v8f acc[4][2];
  #pragma unroll
  for (int i = 0; i < 4; ++i)
    #pragma unroll
    for (int j = 0; j < 2; ++j)
      acc[i][j] = (v8f){0.f, 0.f, 0.f, 0.f, 0.f, 0.f, 0.f, 0.f};

  const int nK = K / BK;
  int buf = 0;

  stage_async(A, W, asBase[0], bsBase[0], tid, K, bm, bn, 0);
  asm volatile("s_wait_asynccnt 0x0" ::: "memory");
  __syncthreads();

  for (int kt = 0; kt < nK; ++kt) {
    if (kt + 1 < nK)
      stage_async(A, W, asBase[buf ^ 1], bsBase[buf ^ 1], tid, K, bm, bn, (kt + 1) * BK);

    // ---- compute: 2 K-steps of 32, 8 WMMAs each ----
    #pragma unroll
    for (int kk = 0; kk < 2; ++kk) {
      v16bf afr[4];
      v16bf bfr[2];
      #pragma unroll
      for (int i = 0; i < 4; ++i) {   // A frag: lane<16 K{0..7,16..23}; lane>=16 K{8..15,24..31}
        int r = wm * 64 + i * 16 + l16;
        const unsigned short* p = &As[buf][r * LDB + kk * 32 + half * 8];
        FragU u;
        u.h[0] = *reinterpret_cast<const us8*>(p);
        u.h[1] = *reinterpret_cast<const us8*>(p + 16);
        afr[i] = u.v;
      }
      #pragma unroll
      for (int j = 0; j < 2; ++j) {   // B frag: lane holds column l16; lane<16 K{0..7,16..23}; lane>=16 K{8..15,24..31}
        int c = wn * 32 + j * 16 + l16;
        const unsigned short* p = &Bs[buf][c * LDB + kk * 32 + half * 8];
        FragU u;
        u.h[0] = *reinterpret_cast<const us8*>(p);
        u.h[1] = *reinterpret_cast<const us8*>(p + 16);
        bfr[j] = u.v;
      }
      #pragma unroll
      for (int i = 0; i < 4; ++i)
        #pragma unroll
        for (int j = 0; j < 2; ++j)
          acc[i][j] = __builtin_amdgcn_wmma_f32_16x16x32_bf16(
              false, afr[i], false, bfr[j], (short)0, acc[i][j], false, false);
    }

    asm volatile("s_wait_asynccnt 0x0" ::: "memory");
    __syncthreads();
    buf ^= 1;
  }

  // ---- epilogue: bias + f32 store. C VGPR e -> M = e + 8*(lane>=16), N = lane&15 ----
  #pragma unroll
  for (int j = 0; j < 2; ++j) {
    int col = bn + wn * 32 + j * 16 + l16;
    float bv = bias[col];
    #pragma unroll
    for (int i = 0; i < 4; ++i) {
      int rbase = bm + wm * 64 + i * 16 + half * 8;
      #pragma unroll
      for (int e = 0; e < 8; ++e)
        C[(size_t)(rbase + e) * Nn + col] = acc[i][j][e] + bv;
    }
  }
}

// Per-token head attention (softmax over heads axis, per reference einsum).
// One thread owns one (token, head) row; writes bf16 output pre-scrambled for
// transpose(0,2,1,3).reshape: dst = b*N*C + h*N*64 + n*64 + d.
__global__ __launch_bounds__(256) void attn_softmax_kernel(
    const float* __restrict__ qkv, unsigned short* __restrict__ scr)
{
  const int t   = blockIdx.x * 256 + threadIdx.x;  // 0 .. 16384*8-1
  const int tok = t >> 3;
  const int h   = t & 7;
  const int b   = tok >> 12;                       // N = 4096
  const int n   = tok & 4095;

  const float* base = qkv + (size_t)tok * 1536;    // (3, H=8, hd=64) per token
  const float* q = base + h * 64;
  const float* k = base + 512;
  const float* v = base + 1024;

  float qr[64];
  #pragma unroll
  for (int d = 0; d < 64; ++d) qr[d] = q[d];

  float s[8];
  #pragma unroll
  for (int g = 0; g < 8; ++g) {
    float acc = 0.f;
    #pragma unroll
    for (int d = 0; d < 64; ++d) acc = fmaf(qr[d], k[g * 64 + d], acc);
    s[g] = acc * 0.125f;                           // hd^-0.5 = 1/8
  }
  float m = s[0];
  #pragma unroll
  for (int g = 1; g < 8; ++g) m = fmaxf(m, s[g]);
  float sum = 0.f;
  #pragma unroll
  for (int g = 0; g < 8; ++g) { s[g] = __expf(s[g] - m); sum += s[g]; }
  float inv = 1.f / sum;
  #pragma unroll
  for (int g = 0; g < 8; ++g) s[g] *= inv;

  unsigned short* o = scr + ((size_t)b * 4096 * 512) + ((size_t)h * 4096 + n) * 64;
  #pragma unroll
  for (int c = 0; c < 4; ++c) {                    // 4 chunks of 16 outputs
    float o16[16];
    #pragma unroll
    for (int i = 0; i < 16; ++i) o16[i] = 0.f;
    #pragma unroll
    for (int g = 0; g < 8; ++g) {
      float pg = s[g];
      #pragma unroll
      for (int i = 0; i < 16; ++i)
        o16[i] = fmaf(pg, v[g * 64 + c * 16 + i], o16[i]);
    }
    #pragma unroll
    for (int i = 0; i < 16; i += 4) {
      us4 hh = { f2bf(o16[i]), f2bf(o16[i + 1]), f2bf(o16[i + 2]), f2bf(o16[i + 3]) };
      *reinterpret_cast<us4*>(o + c * 16 + i) = hh;
    }
  }
}

extern "C" void kernel_launch(void* const* d_in, const int* in_sizes, int n_in,
                              void* d_out, int out_size, void* d_ws, size_t ws_size,
                              hipStream_t stream) {
  const float* x     = (const float*)d_in[0];   // (4, 4096, 512)
  const float* qkv_w = (const float*)d_in[1];   // (1536, 512)
  const float* qkv_b = (const float*)d_in[2];   // (1536,)
  const float* out_w = (const float*)d_in[3];   // (512, 512)
  const float* out_b = (const float*)d_in[4];   // (512,)
  float* out = (float*)d_out;                   // (4, 4096, 512)

  const int M   = 4 * 4096;   // 16384 tokens
  const int C   = 512;
  const int QKV = 1536;

  // ---- workspace layout ----
  char* ws = (char*)d_ws;
  unsigned short* xb  = (unsigned short*)ws;                        // 16 MB
  unsigned short* wqb = xb  + (size_t)M * C;                        // 1.5 MB
  unsigned short* wob = wqb + (size_t)QKV * C;                      // 0.5 MB
  float*          qkv = (float*)(wob + (size_t)C * C);              // 96 MB
  unsigned short* scr = (unsigned short*)(qkv + (size_t)M * QKV);   // 16 MB

  dim3 blk(256);
  // 0) one-pass f32 -> bf16 of activations and weights
  cvt_f32_bf16_kernel<<<dim3((M * C / 4 + 255) / 256), blk, 0, stream>>>(x, xb, M * C / 4);
  cvt_f32_bf16_kernel<<<dim3((QKV * C / 4 + 255) / 256), blk, 0, stream>>>(qkv_w, wqb, QKV * C / 4);
  cvt_f32_bf16_kernel<<<dim3((C * C / 4 + 255) / 256), blk, 0, stream>>>(out_w, wob, C * C / 4);
  // 1) QKV projection: x @ qkv_w^T + qkv_b  (f32 accum, f32 out for softmax)
  gemm_bf16_wmma_kernel<<<dim3(QKV / 128, M / 128), blk, 0, stream>>>(
      xb, wqb, qkv_b, qkv, M, QKV, C);
  // 2) per-token head-softmax attention + scramble (writes bf16)
  attn_softmax_kernel<<<dim3((M * 8) / 256), blk, 0, stream>>>(qkv, scr);
  // 3) output projection: scr @ out_w^T + out_b
  gemm_bf16_wmma_kernel<<<dim3(C / 128, M / 128), blk, 0, stream>>>(
      scr, wob, out_b, out, M, C, C);
}
